// MultiChainReasoner_1400159339037
// MI455X (gfx1250) — matrix operand order repaired and checked by reading, not compiled, not run on hardware
//
#include <hip/hip_runtime.h>
#include <hip/hip_bf16.h>

#define N_NODES 1024
#define DIM     256
#define HID     64
#define NCH     20
#define NSTEP   3
#define NEDGE   16384

typedef __attribute__((ext_vector_type(16))) __bf16 bf16x16;
typedef __attribute__((ext_vector_type(8)))  float  floatx8;

#define WMMA_BF16(a, b, c) \
  __builtin_amdgcn_wmma_f32_16x16x32_bf16(false, (a), false, (b), (short)0, (c), false, false)

__device__ __forceinline__ int lane_id() { return (int)(threadIdx.x & 31); }

// ======================= packed bf16 operand layouts ==========================
// B tile (32x16) for V_WMMA_*_16X16X32: lane l -> N = l&15, K-half = l>>4,
// element e -> K = kt*32 + (l>>4)*16 + e.  Packed: [(kt*ntn + nt)*32 + l][16e]
// A tile (16x32): lane l -> M = l&15, element e -> K = kt*32 + (l>>4)*16 + e
// Packed: [kt*32 + l][16e]. Both give one 32B contiguous load per lane per tile.

__device__ __forceinline__ bf16x16 load_b_pk(const __bf16* base, int ntn, int kt, int nt) {
  return *(const bf16x16*)(base + (((size_t)(kt * ntn + nt)) * 32 + lane_id()) * 16);
}
__device__ __forceinline__ bf16x16 load_a_pk(const __bf16* base, int kt) {
  return *(const bf16x16*)(base + ((size_t)kt * 32 + lane_id()) * 16);
}

// cooperative pack of a 16-row f32 A panel (rows m0..m0+15, K cols) into LDS bf16 A layout
__device__ __forceinline__ void pack_a_lds(const float* __restrict__ A, int ld, int m0,
                                           int K, __bf16* aBuf, int tid, int nth) {
  int total = (K >> 5) * 512;
  for (int i = tid; i < total; i += nth) {
    int e  = i & 15;
    int l  = (i >> 4) & 31;
    int kt = i >> 9;
    int m  = l & 15;
    int k  = kt * 32 + ((l >> 4) << 4) + e;
    aBuf[i] = (__bf16)A[(size_t)(m0 + m) * ld + k];
  }
}

// pack B panel (K x N) from f32 (row-major, or transposed source N x K) into bf16 B layout
__global__ void k_pack_b_batch(const float* __restrict__ src, size_t sstride,
                               __bf16* __restrict__ dst, size_t dstride,
                               int K, int N, int transposed) {
  int c = blockIdx.y;
  const float* s = src + (size_t)c * sstride;
  __bf16* d = dst + (size_t)c * dstride;
  int idx = blockIdx.x * blockDim.x + threadIdx.x;
  if (idx >= K * N) return;
  int e  = idx & 15;
  int l  = (idx >> 4) & 31;
  int t  = idx >> 9;
  int ntn = N >> 4;
  int kt = t / ntn, nt = t - kt * ntn;
  int k  = kt * 32 + ((l >> 4) << 4) + e;
  int n  = nt * 16 + (l & 15);
  float v = transposed ? s[(size_t)n * K + k] : s[(size_t)k * N + n];
  d[idx] = (__bf16)v;
}

// C/D: f32 16x16; vgpr r -> row = (l>>4)*8 + r, col = l&15.
__device__ __forceinline__ void store_c_tile(float* dst, int ld, int m0, int n0,
                                             floatx8 c, const float* bias, int bn0,
                                             int do_relu) {
  const int l   = lane_id();
  const int col = n0 + (l & 15);
  float bv = bias ? bias[bn0 + (l & 15)] : 0.0f;
#pragma unroll
  for (int r = 0; r < 8; ++r) {
    int row = m0 + ((l >> 4) << 3) + r;
    float v = c[r] + bv;
    if (do_relu) v = fmaxf(v, 0.0f);
    dst[(size_t)row * ld + col] = v;
  }
}

// ---------------- mask ----------------
__global__ void k_zero(float* p, int n) {
  int i = blockIdx.x * blockDim.x + threadIdx.x;
  if (i < n) p[i] = 0.0f;
}
__global__ void k_scatter_mask(const int* __restrict__ edges, float* __restrict__ mask, int ne) {
  int i = blockIdx.x * blockDim.x + threadIdx.x;
  if (i < ne) mask[edges[2 * i + 1]] = 1.0f;
}

// ------- generic GEMM: C = act(A @ Bpk + bias); N must be 64 (4 waves x 16 cols) -------
__global__ void __launch_bounds__(128) k_gemm(const float* __restrict__ A,
                                              const __bf16* __restrict__ Bpk,
                                              const float* __restrict__ bias,
                                              float* __restrict__ C,
                                              int M, int K, int N, int do_relu) {
  __shared__ __align__(32) __bf16 aBuf[40 * 512];   // K <= 1280
  int tid = threadIdx.x;
  int m0 = blockIdx.x * 16;
  pack_a_lds(A, K, m0, K, aBuf, tid, 128);
  __syncthreads();
  int wave = tid >> 5;
  floatx8 acc = {};
  for (int kt = 0; kt < (K >> 5); ++kt)
    acc = WMMA_BF16(load_a_pk(aBuf, kt), load_b_pk(Bpk, 4, kt, wave), acc);
  store_c_tile(C, N, m0, wave * 16, acc, bias, wave * 16, do_relu);
}

// ------------- states0[c][n][h] = (ne @ Wci)[n][c*H+h] + bci -------------
__global__ void __launch_bounds__(128) k_states0(const float* __restrict__ ne,
                                                 const __bf16* __restrict__ WciB,
                                                 const float* __restrict__ bci,
                                                 float* __restrict__ states) {
  __shared__ __align__(32) __bf16 aBuf[2 * 512];
  int c = blockIdx.y, tid = threadIdx.x, m0 = blockIdx.x * 16;
  pack_a_lds(ne, HID, m0, HID, aBuf, tid, 128);
  __syncthreads();
  int wave = tid >> 5;
  floatx8 acc = {};
  for (int kt = 0; kt < 2; ++kt)
    acc = WMMA_BF16(load_a_pk(aBuf, kt), load_b_pk(WciB, 80, kt, c * 4 + wave), acc);
  store_c_tile(states + (size_t)c * N_NODES * HID, HID, m0, wave * 16, acc,
               bci, c * HID + wave * 16, 0);
}

// ------------- nodes_part[c] = ne @ w_nodes[c] -------------
__global__ void __launch_bounds__(128) k_nodes_part(const float* __restrict__ ne,
                                                    const __bf16* __restrict__ wnB,
                                                    float* __restrict__ npart) {
  __shared__ __align__(32) __bf16 aBuf[2 * 512];
  int c = blockIdx.y, tid = threadIdx.x, m0 = blockIdx.x * 16;
  pack_a_lds(ne, HID, m0, HID, aBuf, tid, 128);
  __syncthreads();
  int wave = tid >> 5;
  floatx8 acc = {};
  for (int kt = 0; kt < 2; ++kt)
    acc = WMMA_BF16(load_a_pk(aBuf, kt), load_b_pk(wnB + (size_t)c * 4096, 4, kt, wave), acc);
  store_c_tile(npart + (size_t)c * N_NODES * HID, HID, m0, wave * 16, acc, nullptr, 0, 0);
}

// ------------- pre = s@w_state + ba1 ; gh = s@Whh^T + bhh -------------
__global__ void __launch_bounds__(256) k_pre_gh(const float* __restrict__ states,
                                                const __bf16* __restrict__ wstB,
                                                const float* __restrict__ ba1,
                                                const __bf16* __restrict__ whhTB,
                                                const float* __restrict__ bhh,
                                                float* __restrict__ pre,
                                                float* __restrict__ gh) {
  __shared__ __align__(32) __bf16 aBuf[2 * 512];
  int c = blockIdx.y, tid = threadIdx.x, m0 = blockIdx.x * 16;
  const float* s = states + (size_t)c * N_NODES * HID;
  pack_a_lds(s, HID, m0, HID, aBuf, tid, 256);
  __syncthreads();
  int wave = tid >> 5;
  for (int t = wave; t < 16; t += 8) {
    floatx8 acc = {};
    if (t < 4) {
      for (int kt = 0; kt < 2; ++kt)
        acc = WMMA_BF16(load_a_pk(aBuf, kt),
                        load_b_pk(wstB + (size_t)c * 4096, 4, kt, t), acc);
      store_c_tile(pre + (size_t)c * N_NODES * HID, HID, m0, t * 16, acc,
                   ba1 + c * HID, t * 16, 0);
    } else {
      int nt = t - 4;
      for (int kt = 0; kt < 2; ++kt)
        acc = WMMA_BF16(load_a_pk(aBuf, kt),
                        load_b_pk(whhTB + (size_t)c * 12288, 12, kt, nt), acc);
      store_c_tile(gh + (size_t)c * N_NODES * 3 * HID, 3 * HID, m0, nt * 16, acc,
                   bhh + c * 3 * HID, nt * 16, 0);
    }
  }
}

// ------- fused: scores(relu-dot, VALU f32) -> softmax (LDS) -> ctx = attw @ neB (WMMA)
__global__ void __launch_bounds__(256) k_attn_ctx(const float* __restrict__ pre,
                                                  const float* __restrict__ npart,
                                                  const float* __restrict__ Wa2,
                                                  const float* __restrict__ ba2,
                                                  const float* __restrict__ mask,
                                                  const __bf16* __restrict__ neB,
                                                  float* __restrict__ ctx) {
  extern __shared__ __align__(32) char ldsraw[];
  __bf16* aBuf  = (__bf16*)ldsraw;                        // 32*512 bf16 = 32KB
  float*  sc    = (float*)(ldsraw + 32 * 512 * 2);        // 16*1024 f32 = 64KB
  float*  pre_t = sc + 16 * N_NODES;                      // 16*64
  float*  wa2_s = pre_t + 16 * HID;                       // 64
  float*  red   = wa2_s + HID;                            // 256
  float*  rstat = red + 256;                              // 16
  float*  cpart = rstat + 16;                             // 8*16*16

  const int c   = blockIdx.y;
  const int b0  = blockIdx.x * 16;
  const int tid = threadIdx.x;
  const float* prec = pre + ((size_t)c * N_NODES + b0) * HID;
  const float* npc  = npart + (size_t)c * N_NODES * HID;

  for (int i = tid; i < 16 * HID; i += 256) pre_t[i] = prec[i];
  if (tid < HID) wa2_s[tid] = Wa2[c * HID + tid];
  __syncthreads();

  const int b  = tid >> 4;
  const int nl = tid & 15;
  const float ba2c = ba2[c];
  const float4* pr4 = (const float4*)(pre_t + b * HID);
  const float4* w4  = (const float4*)wa2_s;

  // scores[b][n] = (sum_h relu(pre+np)*wa2 + ba2) * mask[n]   -- VALU f32
  for (int chunk = 0; chunk < N_NODES / 16; ++chunk) {
    int n = chunk * 16 + nl;
    const float4* np4 = (const float4*)(npc + (size_t)n * HID);
    if (chunk + 1 < N_NODES / 16)
      __builtin_prefetch(npc + (size_t)(n + 16) * HID, 0, 0);
    float acc = 0.0f;
#pragma unroll
    for (int q = 0; q < HID / 4; ++q) {
      float4 a4 = np4[q], p4 = pr4[q], ww = w4[q];
      acc += fmaxf(p4.x + a4.x, 0.0f) * ww.x;
      acc += fmaxf(p4.y + a4.y, 0.0f) * ww.y;
      acc += fmaxf(p4.z + a4.z, 0.0f) * ww.z;
      acc += fmaxf(p4.w + a4.w, 0.0f) * ww.w;
    }
    sc[b * N_NODES + n] = (acc + ba2c) * mask[n];
  }
  __syncthreads();

  // row max
  float mx = -1e30f;
  for (int n = nl; n < N_NODES; n += 16) mx = fmaxf(mx, sc[b * N_NODES + n]);
  red[tid] = mx;
  __syncthreads();
  if (nl == 0) {
    float mm = red[b * 16];
    for (int i = 1; i < 16; ++i) mm = fmaxf(mm, red[b * 16 + i]);
    rstat[b] = mm;
  }
  __syncthreads();
  float rmax = rstat[b];
  __syncthreads();

  // exp + row sum
  float ssum = 0.0f;
  for (int n = nl; n < N_NODES; n += 16) {
    float e = __expf(sc[b * N_NODES + n] - rmax);
    sc[b * N_NODES + n] = e;
    ssum += e;
  }
  red[tid] = ssum;
  __syncthreads();
  if (nl == 0) {
    float s2 = 0.0f;
    for (int i = 0; i < 16; ++i) s2 += red[b * 16 + i];
    rstat[b] = s2;
  }
  __syncthreads();

  // pack exp-scores into bf16 A layout once (shared by all waves)
  pack_a_lds(sc, N_NODES, 0, N_NODES, aBuf, tid, 256);
  __syncthreads();

  // ctx tile (16x64) = attw(16x1024) @ ne(1024x64); 8 waves = 4 H-tiles x 2 K-halves
  const int wave  = tid >> 5;
  const int htile = wave & 3;
  const int khalf = wave >> 2;
  floatx8 acc = {};
  for (int kt = khalf * 16; kt < khalf * 16 + 16; ++kt)
    acc = WMMA_BF16(load_a_pk(aBuf, kt), load_b_pk(neB, 4, kt, htile), acc);
  {
    const int l = lane_id();
    const int rbase = (l >> 4) << 3;
    const int col = l & 15;
#pragma unroll
    for (int r = 0; r < 8; ++r)
      cpart[((size_t)wave * 16 + rbase + r) * 16 + col] = acc[r];
  }
  __syncthreads();

  // combine K-halves, divide by row sum, store ctx
  {
    const int row = tid >> 4;
    for (int q = 0; q < 4; ++q) {
      int h  = (tid & 15) + q * 16;
      int ht = h >> 4, hc = h & 15;
      float v = cpart[((size_t)ht * 16 + row) * 16 + hc]
              + cpart[((size_t)(4 + ht) * 16 + row) * 16 + hc];
      ctx[((size_t)c * N_NODES + b0 + row) * HID + h] = v / rstat[row];
    }
  }
}

// ------------- GRU: gi = ctx@Wih^T + bih ; s = (1-z)*cand + z*s -------------
__global__ void __launch_bounds__(256) k_gru(const float* __restrict__ ctx,
                                             const float* __restrict__ gh,
                                             const __bf16* __restrict__ wihTB,
                                             const float* __restrict__ bih,
                                             float* __restrict__ states) {
  __shared__ __align__(32) __bf16 aBuf[2 * 512];
  __shared__ float gi_s[16 * 3 * HID];
  int c = blockIdx.y, b0 = blockIdx.x * 16, tid = threadIdx.x;
  pack_a_lds(ctx + (size_t)c * N_NODES * HID, HID, b0, HID, aBuf, tid, 256);
  __syncthreads();

  int wave = tid >> 5;
  for (int t = wave; t < 12; t += 8) {
    floatx8 acc = {};
    for (int kt = 0; kt < 2; ++kt)
      acc = WMMA_BF16(load_a_pk(aBuf, kt),
                      load_b_pk(wihTB + (size_t)c * 12288, 12, kt, t), acc);
    store_c_tile(gi_s, 3 * HID, 0, t * 16, acc, bih + c * 3 * HID, t * 16, 0);
  }
  __syncthreads();

  const float* ghc = gh + ((size_t)c * N_NODES + b0) * 3 * HID;
  float* sp = states + ((size_t)c * N_NODES + b0) * HID;
  for (int i = tid; i < 16 * HID; i += 256) {
    int row = i >> 6, h = i & 63;
    float ir = gi_s[row * 192 + h],        hr  = ghc[row * 192 + h];
    float iz = gi_s[row * 192 + 64 + h],   hz  = ghc[row * 192 + 64 + h];
    float ic = gi_s[row * 192 + 128 + h],  hc2 = ghc[row * 192 + 128 + h];
    float r = 1.0f / (1.0f + __expf(-(ir + hr)));
    float z = 1.0f / (1.0f + __expf(-(iz + hz)));
    float cand = tanhf(ic + r * hc2);
    float so = sp[row * HID + h];
    sp[row * HID + h] = (1.0f - z) * cand + z * so;
  }
}

// ------------- stacked[n][c][h] = states[c][n][h] -------------
__global__ void k_transpose(const float* __restrict__ states, float* __restrict__ stacked) {
  int idx = blockIdx.x * blockDim.x + threadIdx.x;
  if (idx >= NCH * N_NODES * HID) return;
  int h = idx & (HID - 1);
  int n = (idx >> 6) & (N_NODES - 1);
  int c = idx >> 16;
  stacked[((size_t)n * NCH + c) * HID + h] = states[idx];
}

// ------------- attn softmax + pred -------------
__global__ void __launch_bounds__(256) k_head(const float* __restrict__ finalv,
                                              const float* __restrict__ stacked,
                                              const float* __restrict__ Wp,
                                              const float* __restrict__ bp,
                                              float* __restrict__ attn,
                                              float* __restrict__ pred) {
  int n = blockIdx.x * blockDim.x + threadIdx.x;
  if (n >= N_NODES) return;
  float f[HID];
#pragma unroll
  for (int h = 0; h < HID; ++h) f[h] = finalv[n * HID + h];
  float s[NCH];
  float mx = -1e30f;
  for (int c = 0; c < NCH; ++c) {
    const float* st = stacked + ((size_t)n * NCH + c) * HID;
    float d = 0.0f;
#pragma unroll 8
    for (int h = 0; h < HID; ++h) d += f[h] * st[h];
    s[c] = d;
    mx = fmaxf(mx, d);
  }
  float ssum = 0.0f;
  for (int c = 0; c < NCH; ++c) { s[c] = __expf(s[c] - mx); ssum += s[c]; }
  for (int c = 0; c < NCH; ++c) attn[n * NCH + c] = s[c] / ssum;
  float p = bp[0];
#pragma unroll 8
  for (int h = 0; h < HID; ++h) p += f[h] * Wp[h];
  pred[n] = p;
}

extern "C" void kernel_launch(void* const* d_in, const int* in_sizes, int n_in,
                              void* d_out, int out_size, void* d_ws, size_t ws_size,
                              hipStream_t stream) {
  const float* X     = (const float*)d_in[0];
  const int*   edges = (const int*)  d_in[1];
  const float* Wt1   = (const float*)d_in[2];
  const float* bt1   = (const float*)d_in[3];
  const float* Wt2   = (const float*)d_in[4];
  const float* bt2   = (const float*)d_in[5];
  const float* Wci   = (const float*)d_in[6];
  const float* bci   = (const float*)d_in[7];
  const float* Wa1   = (const float*)d_in[8];
  const float* ba1   = (const float*)d_in[9];
  const float* Wa2   = (const float*)d_in[10];
  const float* ba2   = (const float*)d_in[11];
  const float* Wih   = (const float*)d_in[12];
  const float* Whh   = (const float*)d_in[13];
  const float* bih   = (const float*)d_in[14];
  const float* bhh   = (const float*)d_in[15];
  const float* Wag1  = (const float*)d_in[16];
  const float* bag1  = (const float*)d_in[17];
  const float* Wag2  = (const float*)d_in[18];
  const float* bag2  = (const float*)d_in[19];
  const float* Wp    = (const float*)d_in[20];
  const float* bp    = (const float*)d_in[21];

  // ---------------- workspace: f32 region ----------------
  float* ws     = (float*)d_ws;
  float* h1     = ws;                                    // 1024*64
  float* nef    = h1 + N_NODES * HID;                    // 1024*64
  float* maskp  = nef + N_NODES * HID;                   // 1024
  float* states = maskp + N_NODES;                       // 20*1024*64
  float* npart  = states + (size_t)NCH * N_NODES * HID;  // 20*1024*64
  float* pre    = npart + (size_t)NCH * N_NODES * HID;   // 20*1024*64
  float* ghp    = pre + (size_t)NCH * N_NODES * HID;     // 20*1024*192
  float* ctx    = ghp + (size_t)NCH * N_NODES * 3 * HID; // 20*1024*64
  float* fin1   = ctx + (size_t)NCH * N_NODES * HID;     // 1024*64
  float* fend   = fin1 + N_NODES * HID;

  // ---------------- workspace: packed bf16 region ----------------
  __bf16* bws   = (__bf16*)fend;
  __bf16* Wt1B  = bws;                 // 8kt*4nt*512  = 16384
  __bf16* Wt2B  = Wt1B + 16384;        // 2*4*512      = 4096
  __bf16* WciB  = Wt2B + 4096;         // 2*80*512     = 81920
  __bf16* wstB  = WciB + 81920;        // 20 * 2*4*512 = 81920
  __bf16* wnB   = wstB + 81920;        // 20 * 4096    = 81920
  __bf16* whhTB = wnB + 81920;         // 20 * 2*12*512= 245760
  __bf16* wihTB = whhTB + 245760;      // 245760
  __bf16* Wag1B = wihTB + 245760;      // 40*4*512     = 81920
  __bf16* Wag2B = Wag1B + 81920;       // 4096
  __bf16* neB   = Wag2B + 4096;        // 32*4*512     = 65536

  // output layout: stacked | final | attn | pred
  float* out     = (float*)d_out;
  float* stacked = out;
  float* finalp  = out + (size_t)N_NODES * NCH * HID;
  float* attnp   = finalp + (size_t)N_NODES * HID;
  float* predp   = attnp + (size_t)N_NODES * NCH;

  // ---------------- weight packing (one-time per launch) ----------------
  k_pack_b_batch<<<dim3((256*64+255)/256, 1),  256, 0, stream>>>(Wt1, 0, Wt1B, 0, 256, 64, 0);
  k_pack_b_batch<<<dim3((64*64+255)/256, 1),   256, 0, stream>>>(Wt2, 0, Wt2B, 0, 64, 64, 0);
  k_pack_b_batch<<<dim3((64*1280+255)/256, 1), 256, 0, stream>>>(Wci, 0, WciB, 0, 64, 1280, 0);
  k_pack_b_batch<<<dim3((64*64+255)/256, NCH), 256, 0, stream>>>(Wa1, 2*HID*HID, wstB, 4096, 64, 64, 0);
  k_pack_b_batch<<<dim3((64*64+255)/256, NCH), 256, 0, stream>>>(Wa1 + HID*HID, 2*HID*HID, wnB, 4096, 64, 64, 0);
  k_pack_b_batch<<<dim3((64*192+255)/256, NCH),256, 0, stream>>>(Whh, 3*HID*HID, whhTB, 12288, 64, 192, 1);
  k_pack_b_batch<<<dim3((64*192+255)/256, NCH),256, 0, stream>>>(Wih, 3*HID*HID, wihTB, 12288, 64, 192, 1);
  k_pack_b_batch<<<dim3((1280*64+255)/256, 1), 256, 0, stream>>>(Wag1, 0, Wag1B, 0, 1280, 64, 0);
  k_pack_b_batch<<<dim3((64*64+255)/256, 1),   256, 0, stream>>>(Wag2, 0, Wag2B, 0, 64, 64, 0);

  // mask
  k_zero<<<(N_NODES + 255) / 256, 256, 0, stream>>>(maskp, N_NODES);
  k_scatter_mask<<<(NEDGE + 255) / 256, 256, 0, stream>>>(edges, maskp, NEDGE);

  // node embedding + pack ne as B operand
  k_gemm<<<dim3(N_NODES / 16, 1), 128, 0, stream>>>(X, Wt1B, bt1, h1, N_NODES, DIM, HID, 1);
  k_gemm<<<dim3(N_NODES / 16, 1), 128, 0, stream>>>(h1, Wt2B, bt2, nef, N_NODES, HID, HID, 1);
  k_pack_b_batch<<<dim3((1024*64+255)/256, 1), 256, 0, stream>>>(nef, 0, neB, 0, 1024, 64, 0);

  // per-chain precompute
  k_states0<<<dim3(N_NODES / 16, NCH), 128, 0, stream>>>(nef, WciB, bci, states);
  k_nodes_part<<<dim3(N_NODES / 16, NCH), 128, 0, stream>>>(nef, wnB, npart);

  const size_t lds_bytes = (size_t)32 * 512 * 2 +
      (16 * N_NODES + 16 * HID + HID + 256 + 16 + 8 * 16 * 16) * sizeof(float);

  for (int step = 0; step < NSTEP; ++step) {
    k_pre_gh<<<dim3(N_NODES / 16, NCH), 256, 0, stream>>>(states, wstB, ba1, whhTB, bhh, pre, ghp);
    k_attn_ctx<<<dim3(N_NODES / 16, NCH), 256, lds_bytes, stream>>>(pre, npart, Wa2, ba2,
                                                                    maskp, neB, ctx);
    k_gru<<<dim3(N_NODES / 16, NCH), 256, 0, stream>>>(ctx, ghp, wihTB, bih, states);
  }

  // outputs
  k_transpose<<<(NCH * N_NODES * HID + 255) / 256, 256, 0, stream>>>(states, stacked);
  k_gemm<<<dim3(N_NODES / 16, 1), 128, 0, stream>>>(stacked, Wag1B, bag1, fin1,
                                                    N_NODES, NCH * HID, HID, 1);
  k_gemm<<<dim3(N_NODES / 16, 1), 128, 0, stream>>>(fin1, Wag2B, bag2, finalp,
                                                    N_NODES, HID, HID, 0);
  k_head<<<(N_NODES + 255) / 256, 256, 0, stream>>>(finalp, stacked, Wp, bp, attnp, predp);
}
